// PPCrossMultiHeadAttention_28613072126435
// MI455X (gfx1250) — matrix operand order repaired and checked by reading, not compiled, not run on hardware
//
#include <hip/hip_runtime.h>

// ---------------------------------------------------------------------------
// PPCrossMultiHeadAttention for MI455X (gfx1250, wave32, bf16 WMMA)
//   B=4, N=M=2048, D=512, H=8, C=64
//   out = [hidden (B*N*D f32), attn (B*H*N*M f32)]
// ---------------------------------------------------------------------------

constexpr int BB = 4;
constexpr int NN = 2048;
constexpr int MM = 2048;
constexpr int DD = 512;
constexpr int HH = 8;
constexpr int CC = 64;

typedef __bf16 bf16_t;
typedef bf16_t v16bf __attribute__((ext_vector_type(16)));
typedef bf16_t v8bf  __attribute__((ext_vector_type(8)));
typedef float  v8f   __attribute__((ext_vector_type(8)));

static __device__ __forceinline__ unsigned short f2bf(float f) {
    union { __bf16 h; unsigned short u; } c;
    c.h = (__bf16)f;            // hardware cvt, round-to-nearest-even
    return c.u;
}
static __device__ __forceinline__ unsigned pack2(float x, float y) {
    return (unsigned)f2bf(x) | ((unsigned)f2bf(y) << 16);
}

// A-operand (16x32 bf16): lane = (row, kh); 16 contiguous K values per lane at +16*kh
static __device__ __forceinline__ v16bf mk_a(const unsigned short* rowp, int kh) {
    v8bf lo = *reinterpret_cast<const v8bf*>(rowp + 16 * kh);
    v8bf hi = *reinterpret_cast<const v8bf*>(rowp + 16 * kh + 8);
    v16bf a;
#pragma unroll
    for (int j = 0; j < 8; ++j) { a[j] = lo[j]; a[j + 8] = hi[j]; }
    return a;
}

// B-operand (32x16 bf16): lane = (col, kh); 16 contiguous K values per lane
static __device__ __forceinline__ v16bf mk_b(const unsigned short* p) {
    v8bf lo = *reinterpret_cast<const v8bf*>(p);
    v8bf hi = *reinterpret_cast<const v8bf*>(p + 8);
    v16bf b;
#pragma unroll
    for (int j = 0; j < 8; ++j) { b[j] = lo[j]; b[j + 8] = hi[j]; }
    return b;
}

static __device__ __forceinline__ v8f wmma_bf16(v16bf a, v16bf b, v8f c) {
    return __builtin_amdgcn_wmma_f32_16x16x32_bf16(false, a, false, b, (short)0, c,
                                                   false, false);
}

// Async 16-byte copy global -> LDS (CDNA5 ASYNCcnt path, bypasses VGPRs)
static __device__ __forceinline__ void async_cp16(void* lds_dst, const void* gsrc) {
    unsigned           d = (unsigned)reinterpret_cast<size_t>(lds_dst);
    unsigned long long g = (unsigned long long)reinterpret_cast<size_t>(gsrc);
    asm volatile("global_load_async_to_lds_b128 %0, %1, off"
                 :: "v"(d), "v"(g) : "memory");
}
static __device__ __forceinline__ void async_wait0() {
    asm volatile("s_wait_asynccnt 0x0" ::: "memory");
}

// ---------------------------------------------------------------------------
// Projection: Y[r][c] = bf16( X[r][:] . W[:][c] + bias[c] ), R=8192, D=512
// Block tile 128x64, BK=32, 256 threads (8 waves), wave w owns 16-row stripe.
// ---------------------------------------------------------------------------
__global__ __launch_bounds__(256) void proj_kernel(const float* __restrict__ X,
                                                   const float* __restrict__ W,
                                                   const float* __restrict__ bias,
                                                   unsigned short* __restrict__ Y) {
    __shared__ unsigned short Xs[128][32];
    __shared__ unsigned short Wt[64][32];   // transposed: Wt[col][k]
    const int tid  = threadIdx.x;
    const int col0 = blockIdx.x * 64;
    const int row0 = blockIdx.y * 128;
    const int wave = tid >> 5, lane = tid & 31, l15 = lane & 15, kh = lane >> 4;
    const int rt = wave;
    v8f acc[4] = {};

    for (int kt = 0; kt < DD / 32; ++kt) {
        // X tile 128x32 fp32 -> bf16 LDS : 1024 float4, 4 per thread
#pragma unroll
        for (int i = 0; i < 4; ++i) {
            int j = tid * 4 + i;
            int r = j >> 3, ch = j & 7;
            float4 f = *reinterpret_cast<const float4*>(
                X + (size_t)(row0 + r) * DD + kt * 32 + ch * 4);
            uint2 pk; pk.x = pack2(f.x, f.y); pk.y = pack2(f.z, f.w);
            *reinterpret_cast<uint2*>(&Xs[r][ch * 4]) = pk;
        }
        // W tile 32x64 fp32 -> transposed bf16 LDS : 512 float4, 2 per thread
#pragma unroll
        for (int i = 0; i < 2; ++i) {
            int j = tid * 2 + i;
            int k = j >> 4, ch = j & 15;
            float4 f = *reinterpret_cast<const float4*>(
                W + (size_t)(kt * 32 + k) * DD + col0 + ch * 4);
            Wt[ch * 4 + 0][k] = f2bf(f.x);
            Wt[ch * 4 + 1][k] = f2bf(f.y);
            Wt[ch * 4 + 2][k] = f2bf(f.z);
            Wt[ch * 4 + 3][k] = f2bf(f.w);
        }
        __syncthreads();
        v16bf a = mk_a(&Xs[16 * rt + l15][0], kh);
#pragma unroll
        for (int ct = 0; ct < 4; ++ct) {
            v16bf b = mk_b(&Wt[16 * ct + l15][16 * kh]);
            acc[ct] = wmma_bf16(a, b, acc[ct]);
        }
        __syncthreads();
    }
#pragma unroll
    for (int ct = 0; ct < 4; ++ct) {
        int col = col0 + 16 * ct + l15;
        float bv = bias[col];
#pragma unroll
        for (int v = 0; v < 8; ++v) {
            int r = row0 + 16 * rt + v + 8 * kh;
            Y[(size_t)r * DD + col] = f2bf(acc[ct][v] + bv);
        }
    }
}

// ---------------------------------------------------------------------------
// Scores: attn[b,h,n,m] = (q.k + p.r) / 8  (pre-softmax), 64x64 tile per block
// Tiles staged via GLOBAL_LOAD_ASYNC_TO_LDS_B128 (ASYNCcnt).
// ---------------------------------------------------------------------------
__global__ __launch_bounds__(256) void scores_kernel(
    const unsigned short* __restrict__ qw, const unsigned short* __restrict__ kw,
    const unsigned short* __restrict__ pw, const unsigned short* __restrict__ rw,
    float* __restrict__ attn) {
    __shared__ unsigned short qs[64][64], ks[64][64], ps[64][64], rs[64][64];
    const int tid = threadIdx.x;
    const int bh = blockIdx.z;
    const int b = bh >> 3, h = bh & 7;
    const int n0 = blockIdx.y * 64, m0 = blockIdx.x * 64;

    const unsigned short* qsrc = qw + ((size_t)(b * NN + n0)) * DD + h * CC;
    const unsigned short* psrc = pw + ((size_t)(b * NN + n0)) * DD + h * CC;
    const unsigned short* ksrc = kw + ((size_t)(b * MM + m0)) * DD + h * CC;
    const unsigned short* rsrc = rw + ((size_t)(b * MM + m0)) * DD + h * CC;
#pragma unroll
    for (int i = 0; i < 2; ++i) {
        int j = tid * 2 + i;
        int r = j >> 3, ch = j & 7;
        size_t so = (size_t)r * DD + ch * 8;
        async_cp16(&qs[r][ch * 8], qsrc + so);
        async_cp16(&ps[r][ch * 8], psrc + so);
        async_cp16(&ks[r][ch * 8], ksrc + so);
        async_cp16(&rs[r][ch * 8], rsrc + so);
    }
    async_wait0();
    __syncthreads();

    const int wave = tid >> 5, lane = tid & 31, l15 = lane & 15, kh = lane >> 4;
    const int rt = wave & 3;
    const int row = 16 * rt + l15;

    v16bf aq0 = mk_a(&qs[row][0], kh);    // K = 0..31
    v16bf aq1 = mk_a(&qs[row][32], kh);   // K = 32..63
    v16bf ap0 = mk_a(&ps[row][0], kh);
    v16bf ap1 = mk_a(&ps[row][32], kh);

#pragma unroll
    for (int i = 0; i < 2; ++i) {
        int ct = (wave >> 2) * 2 + i;
        int mc = 16 * ct + l15;   // key row within tile == score column
        v16bf bk0 = mk_b(&ks[mc][16 * kh]);
        v16bf bk1 = mk_b(&ks[mc][32 + 16 * kh]);
        v16bf br0 = mk_b(&rs[mc][16 * kh]);
        v16bf br1 = mk_b(&rs[mc][32 + 16 * kh]);
        v8f acc = {};
        acc = wmma_bf16(aq0, bk0, acc);
        acc = wmma_bf16(aq1, bk1, acc);
        acc = wmma_bf16(ap0, br0, acc);
        acc = wmma_bf16(ap1, br1, acc);

        float* out = attn + ((size_t)bh * NN + (n0 + 16 * rt)) * MM + m0 + 16 * ct;
#pragma unroll
        for (int v = 0; v < 8; ++v) {
            out[(size_t)(v + 8 * kh) * MM + l15] = 0.125f * acc[v];
        }
    }
}

// ---------------------------------------------------------------------------
// Softmax over last dim (M=2048): one wave per row, register-resident
// ---------------------------------------------------------------------------
__global__ __launch_bounds__(256) void softmax_kernel(float* __restrict__ attn) {
    const int wave = threadIdx.x >> 5, lane = threadIdx.x & 31;
    const size_t row = (size_t)blockIdx.x * 8 + wave;
    float* p = attn + row * MM;

    float4 regs[16];
    float mx = -3.4e38f;
#pragma unroll
    for (int i = 0; i < 16; ++i) {
        regs[i] = *reinterpret_cast<const float4*>(p + (size_t)(lane + 32 * i) * 4);
        mx = fmaxf(mx, fmaxf(fmaxf(regs[i].x, regs[i].y), fmaxf(regs[i].z, regs[i].w)));
    }
#pragma unroll
    for (int off = 16; off >= 1; off >>= 1) mx = fmaxf(mx, __shfl_xor(mx, off, 32));

    float s = 0.f;
#pragma unroll
    for (int i = 0; i < 16; ++i) {
        regs[i].x = __expf(regs[i].x - mx);
        regs[i].y = __expf(regs[i].y - mx);
        regs[i].z = __expf(regs[i].z - mx);
        regs[i].w = __expf(regs[i].w - mx);
        s += regs[i].x + regs[i].y + regs[i].z + regs[i].w;
    }
#pragma unroll
    for (int off = 16; off >= 1; off >>= 1) s += __shfl_xor(s, off, 32);
    float inv = 1.f / s;
#pragma unroll
    for (int i = 0; i < 16; ++i) {
        regs[i].x *= inv; regs[i].y *= inv; regs[i].z *= inv; regs[i].w *= inv;
        *reinterpret_cast<float4*>(p + (size_t)(lane + 32 * i) * 4) = regs[i];
    }
}

// ---------------------------------------------------------------------------
// hidden = attn @ V : per (b,h), 64 rows x 64 cols per block, K-loop over M
// ---------------------------------------------------------------------------
__global__ __launch_bounds__(256) void av_kernel(const float* __restrict__ attn,
                                                 const unsigned short* __restrict__ vw,
                                                 float* __restrict__ hidden) {
    __shared__ unsigned short As[64][32];
    __shared__ unsigned short Vt[64][32];   // Vt[col][k]
    const int tid = threadIdx.x;
    const int bh = blockIdx.y;
    const int b = bh >> 3, h = bh & 7;
    const int n0 = blockIdx.x * 64;
    const int wave = tid >> 5, lane = tid & 31, l15 = lane & 15, kh = lane >> 4;
    const int rt = wave & 3;
    const int ct0 = (wave >> 2) * 2;
    v8f acc[2] = {};

    const float* arow = attn + ((size_t)bh * NN + n0) * MM;
    const unsigned short* vbase = vw + ((size_t)b * MM) * DD + h * CC;

    for (int kt = 0; kt < MM / 32; ++kt) {
        int mb = kt * 32;
        // attn tile 64x32 fp32 -> bf16 LDS : 512 float4, 2 per thread
#pragma unroll
        for (int i = 0; i < 2; ++i) {
            int j = tid * 2 + i;
            int r = j >> 3, ch = j & 7;
            float4 f = *reinterpret_cast<const float4*>(arow + (size_t)r * MM + mb + ch * 4);
            uint2 pk; pk.x = pack2(f.x, f.y); pk.y = pack2(f.z, f.w);
            *reinterpret_cast<uint2*>(&As[r][ch * 4]) = pk;
        }
        // V tile 32x64 bf16 -> transposed LDS : 256 uint4, 1 per thread
        {
            int k = tid >> 3, ch = tid & 7;
            uint4 u = *reinterpret_cast<const uint4*>(vbase + (size_t)(mb + k) * DD + ch * 8);
            unsigned short tmp[8];
            *reinterpret_cast<uint4*>(tmp) = u;
#pragma unroll
            for (int i = 0; i < 8; ++i) Vt[ch * 8 + i][k] = tmp[i];
        }
        // prefetch next K-tile (lowers to global_prefetch_b8)
        if (kt + 1 < MM / 32) {
            __builtin_prefetch(arow + (size_t)(tid >> 2) * MM + mb + 32, 0, 1);
            __builtin_prefetch(vbase + (size_t)(mb + 32 + (tid & 31)) * DD, 0, 1);
        }
        __syncthreads();
        v16bf a = mk_a(&As[16 * rt + l15][0], kh);
#pragma unroll
        for (int i = 0; i < 2; ++i) {
            v16bf bop = mk_b(&Vt[16 * (ct0 + i) + l15][16 * kh]);
            acc[i] = wmma_bf16(a, bop, acc[i]);
        }
        __syncthreads();
    }
#pragma unroll
    for (int i = 0; i < 2; ++i) {
        int col = h * CC + 16 * (ct0 + i) + l15;
#pragma unroll
        for (int v = 0; v < 8; ++v) {
            int r = n0 + 16 * rt + v + 8 * kh;
            hidden[((size_t)b * NN + r) * DD + col] = acc[i][v];
        }
    }
}

// ---------------------------------------------------------------------------
extern "C" void kernel_launch(void* const* d_in, const int* in_sizes, int n_in,
                              void* d_out, int out_size, void* d_ws, size_t ws_size,
                              hipStream_t stream) {
    (void)in_sizes; (void)n_in; (void)out_size; (void)ws_size;

    const float* input_q = (const float*)d_in[0];
    const float* input_k = (const float*)d_in[1];
    const float* input_v = (const float*)d_in[2];
    const float* embed_q = (const float*)d_in[3];
    const float* embed_k = (const float*)d_in[4];
    const float* Wq = (const float*)d_in[5];  const float* bq = (const float*)d_in[6];
    const float* Wk = (const float*)d_in[7];  const float* bk = (const float*)d_in[8];
    const float* Wv = (const float*)d_in[9];  const float* bv = (const float*)d_in[10];
    const float* Wp = (const float*)d_in[11]; const float* bp = (const float*)d_in[12];

    const size_t proj_elems = (size_t)BB * NN * DD;   // 4,194,304 bf16 each
    unsigned short* qws = (unsigned short*)d_ws;
    unsigned short* kws = qws + proj_elems;
    unsigned short* vws = kws + proj_elems;
    unsigned short* pws = vws + proj_elems;
    unsigned short* rws = pws + proj_elems;           // total 40 MB of d_ws

    float* hidden = (float*)d_out;
    float* attn = hidden + (size_t)BB * NN * DD;

    dim3 pg(DD / 64, (BB * NN) / 128);                // (8, 64)
    proj_kernel<<<pg, 256, 0, stream>>>(input_q, Wq, bq, qws);
    proj_kernel<<<pg, 256, 0, stream>>>(input_k, Wk, bk, kws);
    proj_kernel<<<pg, 256, 0, stream>>>(input_v, Wv, bv, vws);
    proj_kernel<<<pg, 256, 0, stream>>>(embed_q, Wp, bp, pws);
    proj_kernel<<<pg, 256, 0, stream>>>(embed_k, Wp, bp, rws);

    scores_kernel<<<dim3(MM / 64, NN / 64, BB * HH), 256, 0, stream>>>(qws, kws, pws,
                                                                       rws, attn);
    softmax_kernel<<<(BB * HH * NN) / 8, 256, 0, stream>>>(attn);
    av_kernel<<<dim3(NN / 64, BB * HH), 256, 0, stream>>>(attn, vws, hidden);
}
